// MultiHeadAttention_901943132789
// MI455X (gfx1250) — compile-verified, matrix operand order checked by reading, not run on hardware
//
#include <hip/hip_runtime.h>

#define D_MODEL 1024
#define NUM_HEADS 16
#define HEAD_DIM 64
#define BATCH 4
#define SEQ 2048

typedef __attribute__((ext_vector_type(16))) _Float16 v16h;
typedef __attribute__((ext_vector_type(8)))  float    v8f;
typedef _Float16 f16;

// ---------------- CDNA5 async global->LDS path (guarded) ----------------
#if defined(__has_builtin)
#if __has_builtin(__builtin_amdgcn_global_load_async_to_lds_b128)
#define HAVE_ASYNC_LDS 1
#endif
#endif
#ifndef HAVE_ASYNC_LDS
#define HAVE_ASYNC_LDS 0
#endif

static __device__ __forceinline__ void cp_glds_b128(f16* lds_dst,
                                                    const f16* gsrc) {
#if HAVE_ASYNC_LDS
  typedef int v4i __attribute__((vector_size(16)));
  __builtin_amdgcn_global_load_async_to_lds_b128(
      (__attribute__((address_space(1))) v4i*)gsrc,
      (__attribute__((address_space(3))) v4i*)lds_dst, 0, 0);
#else
  *(int4*)lds_dst = *(const int4*)gsrc;  // sync fallback via VGPRs
#endif
}

#if HAVE_ASYNC_LDS
#if __has_builtin(__builtin_amdgcn_s_wait_asynccnt)
#define WAIT_ASYNC(n) __builtin_amdgcn_s_wait_asynccnt(n)
#else
#define WAIT_ASYNC(n) asm volatile("s_wait_asynccnt " #n ::: "memory")
#endif
#else
#define WAIT_ASYNC(n)
#endif

// ---- WMMA helper: D = A(16x32 f16) * B(32x16 f16) + C(f32) ----
static __device__ __forceinline__ v8f wmma16(v16h a, v16h b, v8f c) {
  return __builtin_amdgcn_wmma_f32_16x16x32_f16(false, a, false, b, (short)0, c,
                                                false, false);
}

// Fragment loaders per CDNA5 16-bit 16x32 layout (kbase = k0 + hg*8):
static __device__ __forceinline__ v16h frag_from_f32(const float* __restrict__ row,
                                                     int kbase) {
  v16h f;
#pragma unroll
  for (int i = 0; i < 8; ++i) f[i] = (f16)row[kbase + i];
#pragma unroll
  for (int i = 0; i < 8; ++i) f[8 + i] = (f16)row[kbase + 16 + i];
  return f;
}

static __device__ __forceinline__ v16h frag_from_f16(const f16* __restrict__ row,
                                                     int kbase) {
  v16h f;
#pragma unroll
  for (int i = 0; i < 8; ++i) f[i] = row[kbase + i];
#pragma unroll
  for (int i = 0; i < 8; ++i) f[8 + i] = row[kbase + 16 + i];
  return f;
}

// ---------------------------------------------------------------------------
// Kernel 1: QKV projection.  out = X @ W^T + bias, cast f16, head-split.
// One wave computes a 64x32 output tile (4x2 WMMA tiles, 8 WMMAs/k-step).
// transposed==0 -> out[b][h][s][d] (Q,K);  ==1 -> out[b][h][d][s] (V).
// ---------------------------------------------------------------------------
__global__ void __launch_bounds__(256) qkv_proj_kernel(
    const float* __restrict__ X, const float* __restrict__ W,
    const float* __restrict__ bias, f16* __restrict__ out, int transposed) {
  const int lane   = threadIdx.x & 31;
  const int hg     = lane >> 4;
  const int lane16 = lane & 15;
  const int wave   = blockIdx.x * (blockDim.x >> 5) + (threadIdx.x >> 5);
  const int NT_N   = D_MODEL / 32;  // 32 tiles in N
  const int m0     = (wave / NT_N) * 64;
  const int n0     = (wave % NT_N) * 32;

  v8f acc[4][2];
#pragma unroll
  for (int i = 0; i < 4; ++i)
#pragma unroll
    for (int j = 0; j < 2; ++j) acc[i][j] = {};

  const float* arow[4];
#pragma unroll
  for (int i = 0; i < 4; ++i)
    arow[i] = X + (size_t)(m0 + i * 16 + lane16) * D_MODEL;
  const float* brow0 = W + (size_t)(n0 + lane16) * D_MODEL;
  const float* brow1 = W + (size_t)(n0 + 16 + lane16) * D_MODEL;

  for (int k0 = 0; k0 < D_MODEL; k0 += 32) {
    const int kbase = k0 + hg * 8;
    __builtin_prefetch(arow[0] + kbase + 64, 0, 0);
    __builtin_prefetch(brow0 + kbase + 64, 0, 0);
    v16h b0 = frag_from_f32(brow0, kbase);
    v16h b1 = frag_from_f32(brow1, kbase);
#pragma unroll
    for (int i = 0; i < 4; ++i) {
      v16h a = frag_from_f32(arow[i], kbase);
      acc[i][0] = wmma16(a, b0, acc[i][0]);
      acc[i][1] = wmma16(a, b1, acc[i][1]);
    }
  }

#pragma unroll
  for (int am = 0; am < 4; ++am) {
#pragma unroll
    for (int bn = 0; bn < 2; ++bn) {
      const int n    = n0 + bn * 16 + lane16;
      const float bv = bias[n];
      const int h    = n >> 6;
      const int d    = n & 63;
#pragma unroll
      for (int r = 0; r < 8; ++r) {
        const int m = m0 + am * 16 + hg * 8 + r;
        const int b = m >> 11;          // / SEQ
        const int s = m & (SEQ - 1);
        const float val = acc[am][bn][r] + bv;
        size_t idx;
        if (!transposed)
          idx = (((size_t)(b * NUM_HEADS + h) * SEQ) + s) * HEAD_DIM + d;
        else
          idx = (((size_t)(b * NUM_HEADS + h) * HEAD_DIM) + d) * SEQ + s;
        out[idx] = (f16)val;
      }
    }
  }
}

// ---------------------------------------------------------------------------
// Kernel 2: flash attention, workgroup-cooperative.
// Block = 8 waves = 128 query rows of one (b,h).  K/V tiles (32 keys) are
// double-buffered in LDS via async global->LDS loads; each wave runs online
// softmax over its 16 rows with 4 score WMMAs + 4 P*V WMMAs per tile.
// ---------------------------------------------------------------------------
__global__ void __launch_bounds__(256) attn_kernel(
    const f16* __restrict__ Q, const f16* __restrict__ K,
    const f16* __restrict__ Vt, const float* __restrict__ mask,
    f16* __restrict__ AO) {
  __shared__ f16 Kt[2][32][HEAD_DIM];   // [buf][key][d]     8 KB
  __shared__ f16 Vs[2][HEAD_DIM][32];   // [buf][d][key]     8 KB
  __shared__ f16 pbuf[8][16][32];       // per-wave P tiles  8 KB

  const int tid    = threadIdx.x;
  const int lane   = tid & 31;
  const int hg     = lane >> 4;
  const int lane16 = lane & 15;
  const int warp   = tid >> 5;

  const int blk  = blockIdx.x;     // B*H*(S/128) = 1024
  const int qblk = blk & 15;       // 16 chunks of 128 queries
  const int bh   = blk >> 4;       // 0..63
  const int b    = bh >> 4;
  const int h    = bh & 15;
  const int q0   = qblk * 128 + warp * 16;

  const size_t head = (size_t)bh * SEQ * HEAD_DIM;
  const f16* Qh = Q + head;
  const f16* Kh = K + head;
  const f16* Vh = Vt + head;                       // [64][SEQ]
  const float* Mb = mask + (size_t)b * SEQ * SEQ;  // [S][S]

  // cooperative K/V tile fetch: 256 threads x 16B each per tile
  const int kr  = tid >> 3, kc = (tid & 7) * 8;    // K: 32 rows x 64
  const int vd  = tid >> 2, vc = (tid & 3) * 8;    // V: 64 rows x 32
#define ISSUE_TILE(J, BUF)                                                  \
  do {                                                                      \
    cp_glds_b128(&Kt[BUF][kr][kc], Kh + (size_t)((J) + kr) * HEAD_DIM + kc);\
    cp_glds_b128(&Vs[BUF][vd][vc], Vh + (size_t)vd * SEQ + (J) + vc);       \
  } while (0)

  ISSUE_TILE(0, 0);
  ISSUE_TILE(32, 1);

  // Q fragments (reused for all key tiles)
  const f16* qrow = Qh + (size_t)(q0 + lane16) * HEAD_DIM;
  const v16h qa0 = frag_from_f16(qrow, hg * 8);
  const v16h qa1 = frag_from_f16(qrow, 32 + hg * 8);

  const float scale = 0.125f;  // 1/sqrt(HEAD_DIM)

  v8f oacc[4];
#pragma unroll
  for (int c = 0; c < 4; ++c) oacc[c] = {};
  float mrow[8], lrow[8];
#pragma unroll
  for (int r = 0; r < 8; ++r) { mrow[r] = -1e30f; lrow[r] = 0.0f; }

  for (int j = 0; j < SEQ; j += 32) {
    const int buf = (j >> 5) & 1;
    WAIT_ASYNC(2);      // own wave's loads for THIS tile done (older of 2)
    __syncthreads();    // all waves' pieces of the tile are in LDS

    // ---- scores for keys [j, j+32) from LDS K tile ----
    v8f sa = {}, sb = {};
    {
      const f16* kr_a = &Kt[buf][lane16][0];
      const f16* kr_b = &Kt[buf][16 + lane16][0];
      sa = wmma16(qa0, frag_from_f16(kr_a, hg * 8), sa);
      sa = wmma16(qa1, frag_from_f16(kr_a, 32 + hg * 8), sa);
      sb = wmma16(qa0, frag_from_f16(kr_b, hg * 8), sb);
      sb = wmma16(qa1, frag_from_f16(kr_b, 32 + hg * 8), sb);
    }

    // ---- online softmax over the 16-lane C-layout row groups ----
    float corr[8];
#pragma unroll
    for (int r = 0; r < 8; ++r) {
      const int t = q0 + hg * 8 + r;
      const float* mrow_p = Mb + (size_t)t * SEQ + j;
      const float va = sa[r] * scale + mrow_p[lane16];
      const float vb = sb[r] * scale + mrow_p[16 + lane16];
      float tmax = fmaxf(va, vb);
      tmax = fmaxf(tmax, __shfl_xor(tmax, 1, 32));
      tmax = fmaxf(tmax, __shfl_xor(tmax, 2, 32));
      tmax = fmaxf(tmax, __shfl_xor(tmax, 4, 32));
      tmax = fmaxf(tmax, __shfl_xor(tmax, 8, 32));
      const float mnew = fmaxf(mrow[r], tmax);
      const float c  = __expf(mrow[r] - mnew);
      const float pa = __expf(va - mnew);
      const float pb = __expf(vb - mnew);
      float ps = pa + pb;
      ps += __shfl_xor(ps, 1, 32);
      ps += __shfl_xor(ps, 2, 32);
      ps += __shfl_xor(ps, 4, 32);
      ps += __shfl_xor(ps, 8, 32);
      lrow[r] = lrow[r] * c + ps;
      mrow[r] = mnew;
      corr[r] = c;
      pbuf[warp][hg * 8 + r][lane16]      = (f16)pa;
      pbuf[warp][hg * 8 + r][16 + lane16] = (f16)pb;
    }
#pragma unroll
    for (int c = 0; c < 4; ++c)
#pragma unroll
      for (int r = 0; r < 8; ++r) oacc[c][r] *= corr[r];

    __syncthreads();  // P staged in LDS

    // ---- P (16x32) as A-fragment; P*V into 4 chunks of 16 cols ----
    const v16h pf = frag_from_f16(&pbuf[warp][lane16][0], hg * 8);
#pragma unroll
    for (int c = 0; c < 4; ++c) {
      v16h vf = frag_from_f16(&Vs[buf][c * 16 + lane16][0], hg * 8);
      oacc[c] = wmma16(pf, vf, oacc[c]);
    }

    __syncthreads();  // everyone done with buf; safe to refill it
    if (j + 64 < SEQ) ISSUE_TILE(j + 64, buf);
  }
#undef ISSUE_TILE

  // ---- normalize, store AO[b][s][h*64 + d] ----
  float inv[8];
#pragma unroll
  for (int r = 0; r < 8; ++r) inv[r] = 1.0f / lrow[r];
#pragma unroll
  for (int c = 0; c < 4; ++c) {
#pragma unroll
    for (int r = 0; r < 8; ++r) {
      const int t = q0 + hg * 8 + r;
      const size_t idx =
          ((size_t)b * SEQ + t) * D_MODEL + h * HEAD_DIM + c * 16 + lane16;
      AO[idx] = (f16)(oacc[c][r] * inv[r]);
    }
  }
}

// ---------------------------------------------------------------------------
// Kernel 3: output projection.  out = AO(f16) @ w_o^T + b_o  (f32 result)
// 64x32 tile per wave.
// ---------------------------------------------------------------------------
__global__ void __launch_bounds__(256) out_proj_kernel(
    const f16* __restrict__ X, const float* __restrict__ W,
    const float* __restrict__ bias, float* __restrict__ out) {
  const int lane   = threadIdx.x & 31;
  const int hg     = lane >> 4;
  const int lane16 = lane & 15;
  const int wave   = blockIdx.x * (blockDim.x >> 5) + (threadIdx.x >> 5);
  const int NT_N   = D_MODEL / 32;
  const int m0     = (wave / NT_N) * 64;
  const int n0     = (wave % NT_N) * 32;

  v8f acc[4][2];
#pragma unroll
  for (int i = 0; i < 4; ++i)
#pragma unroll
    for (int j = 0; j < 2; ++j) acc[i][j] = {};

  const f16* arow[4];
#pragma unroll
  for (int i = 0; i < 4; ++i)
    arow[i] = X + (size_t)(m0 + i * 16 + lane16) * D_MODEL;
  const float* brow0 = W + (size_t)(n0 + lane16) * D_MODEL;
  const float* brow1 = W + (size_t)(n0 + 16 + lane16) * D_MODEL;

  for (int k0 = 0; k0 < D_MODEL; k0 += 32) {
    const int kbase = k0 + hg * 8;
    v16h b0 = frag_from_f32(brow0, kbase);
    v16h b1 = frag_from_f32(brow1, kbase);
#pragma unroll
    for (int i = 0; i < 4; ++i) {
      v16h a = frag_from_f16(arow[i], kbase);
      acc[i][0] = wmma16(a, b0, acc[i][0]);
      acc[i][1] = wmma16(a, b1, acc[i][1]);
    }
  }

#pragma unroll
  for (int am = 0; am < 4; ++am) {
#pragma unroll
    for (int bn = 0; bn < 2; ++bn) {
      const int n    = n0 + bn * 16 + lane16;
      const float bv = bias[n];
#pragma unroll
      for (int r = 0; r < 8; ++r) {
        const int m = m0 + am * 16 + hg * 8 + r;
        out[(size_t)m * D_MODEL + n] = acc[am][bn][r] + bv;
      }
    }
  }
}

// ---------------------------------------------------------------------------
extern "C" void kernel_launch(void* const* d_in, const int* in_sizes, int n_in,
                              void* d_out, int out_size, void* d_ws,
                              size_t ws_size, hipStream_t stream) {
  (void)in_sizes; (void)n_in; (void)out_size; (void)ws_size;
  const float* q    = (const float*)d_in[0];
  const float* k    = (const float*)d_in[1];
  const float* v    = (const float*)d_in[2];
  const float* mask = (const float*)d_in[3];
  const float* w_q  = (const float*)d_in[4];
  const float* b_q  = (const float*)d_in[5];
  const float* w_k  = (const float*)d_in[6];
  const float* b_k  = (const float*)d_in[7];
  const float* w_v  = (const float*)d_in[8];
  const float* b_v  = (const float*)d_in[9];
  const float* w_o  = (const float*)d_in[10];
  const float* b_o  = (const float*)d_in[11];
  float* out = (float*)d_out;

  const size_t HSZ = (size_t)BATCH * NUM_HEADS * SEQ * HEAD_DIM;  // 8.4M elems
  f16* qh = (f16*)d_ws;
  f16* kh = qh + HSZ;
  f16* vt = kh + HSZ;
  f16* ao = vt + HSZ;

  // (B*S/64) * (D/32) = 4096 waves -> 512 blocks of 8 waves
  const dim3 gGemm(512), blk(256);
  qkv_proj_kernel<<<gGemm, blk, 0, stream>>>(q, w_q, b_q, qh, 0);
  qkv_proj_kernel<<<gGemm, blk, 0, stream>>>(k, w_k, b_k, kh, 0);
  qkv_proj_kernel<<<gGemm, blk, 0, stream>>>(v, w_v, b_v, vt, 1);

  // B*H*(S/128) = 1024 blocks, 8 waves each
  attn_kernel<<<dim3(1024), blk, 0, stream>>>(qh, kh, vt, mask, ao);

  out_proj_kernel<<<gGemm, blk, 0, stream>>>(ao, w_o, b_o, out);
}
